// FRAPModel_82085414961910
// MI455X (gfx1250) — matrix operand order repaired and checked by reading, not compiled, not run on hardware
//
#include <hip/hip_runtime.h>

// FRAP model: B=131072, M=12, IN_DIM=12, P=8, EMB=16
#define BM 12
#define PD 8
#define ED 16
#define WAVES 4
#define THREADS (WAVES * 32)
#define B_PER_BLOCK (WAVES * 16)

typedef __attribute__((ext_vector_type(16))) __bf16 v16bf;
typedef __attribute__((ext_vector_type(8)))  float  v8f;

__device__ __forceinline__ float leaky(float x) { return __builtin_fmaxf(x, 0.01f * x); }

// Zero-instruction wave-local ordering. DS ops from one wave execute in order
// (ISA ch.11); we only need to stop compiler/scheduler reordering around the
// LDS transpose staging, which is wave-private.
__device__ __forceinline__ void wave_fence() {
  asm volatile("" ::: "memory");
  __builtin_amdgcn_wave_barrier();
  asm volatile("" ::: "memory");
}

// B-matrix fragment (K x 16 bf16, K padded to 32) from row-major W[O][C]:
// value = W[nBase+N][K]. Layout (ISA 7.12.2 dense 16-bit B): lane L holds
// column N = L%16; halves e=0..15 hold K = (e/8)*16 + (L/16)*8 + e%8.
__device__ __forceinline__ v16bf load_bfrag(const float* __restrict__ W,
                                            int O, int C, int nBase, int lane) {
  v16bf f;
  const int N = lane & 15;
  const int kBase = (lane >> 4) << 3;
  const int o = nBase + N;
#pragma unroll
  for (int e = 0; e < 16; ++e) {
    int K = ((e >> 3) << 4) + kBase + (e & 7);
    float v = (o < O && K < C) ? W[o * C + K] : 0.0f;
    f[e] = (__bf16)v;
  }
  return f;
}

#define WMMA_BF16(A, Bf, C) \
  __builtin_amdgcn_wmma_f32_16x16x32_bf16(false, (A), false, (Bf), (short)0, (C), false, false)

__global__ void __launch_bounds__(THREADS)
frap_fused_kernel(const float* __restrict__ stat, const int* __restrict__ phase,
                  const int* __restrict__ mask,
                  const float* __restrict__ W1,  const float* __restrict__ b1,
                  const float* __restrict__ emb,
                  const float* __restrict__ We1, const float* __restrict__ be1,
                  const float* __restrict__ We2, const float* __restrict__ be2,
                  const float* __restrict__ Wc1, const float* __restrict__ bc1,
                  const float* __restrict__ Wc2, const float* __restrict__ bc2,
                  const float* __restrict__ Wc3, const float* __restrict__ bc3,
                  float* __restrict__ out)
{
  // Small bias/lookup tables (fp32) — weights themselves live in WMMA B-frags.
  __shared__ float sb1v[12], sEmb[16], sbe1[32], sbe2[16];
  __shared__ float sMask[96], sbc1[20], sbc2[20];
  __shared__ int   pBuf[WAVES][192];            // phase per (b,m) row
  __shared__ __bf16 eBuf[WAVES][192][ED];       // per-(b,m) embeddings
  __shared__ __bf16 encB[WAVES][16][PD][ED];    // pooled encodings
  __shared__ __bf16 xs1[WAVES][16][32];         // D->A transpose staging
  __shared__ __bf16 xs2[WAVES][16][32];

  const int tid  = threadIdx.x;
  const int wave = tid >> 5;
  const int lane = tid & 31;

  if (tid < 12) sb1v[tid] = b1[tid];
  if (tid < 16) { sEmb[tid] = emb[tid]; sbe2[tid] = be2[tid]; }
  if (tid < 32) sbe1[tid] = be1[tid];
  if (tid < 96) sMask[tid] = (mask[tid] > 0) ? 1.0f : 0.0f;
  if (tid < 20) { sbc1[tid] = bc1[tid]; sbc2[tid] = bc2[tid]; }
  __syncthreads();   // only真 cross-wave sync in the kernel

  const long long bBase   = (long long)blockIdx.x * B_PER_BLOCK + wave * 16;
  const long long rowBase = bBase * BM;          // first (b,m) row of this wave

  // phase preload for this wave's 192 rows (coalesced)
  for (int r = lane; r < 192; r += 32) pBuf[wave][r] = phase[rowBase + r];

  // zero xs1 cols 16..31 once (K=16..31 of stage-A GEMM2 A-frags must be 0)
  {
    int zr = lane >> 1, zc = 16 + (lane & 1) * 8;
#pragma unroll
    for (int i = 0; i < 8; ++i) xs1[wave][zr][zc + i] = (__bf16)0.0f;
  }
  wave_fence();

  // ---- lane geometry (ISA 7.12.2) ----
  const int Mrow = lane & 15;            // A-frag row / D-frag column group
  const int base = (lane >> 4) << 3;     // A-frag K base: 0 or 8
  const int N    = lane & 15;            // D-frag column
  const int rOff = base;                 // D-frag row offset: M = r + rOff

  const v8f zero = {0.f, 0.f, 0.f, 0.f, 0.f, 0.f, 0.f, 0.f};

  // =========== Stage A: encoder chain via WMMA (12 tiles of 16 rows) ==========
  const v16bf BW1   = load_bfrag(W1,  12, 12, 0,  lane);
  const v16bf BWe1a = load_bfrag(We1, 32, 16, 0,  lane);
  const v16bf BWe1b = load_bfrag(We1, 32, 16, 16, lane);
  const v16bf BWe2  = load_bfrag(We2, 16, 32, 0,  lane);

  const float esMask = (N < 12) ? 1.0f : 0.0f;   // cols 0..11 = es, 12..15 = emb
  const int   b1Idx  = (N < 12) ? N : 11;
  const int   embCol = (N >= 12) ? (N - 12) : 0;
  const float hi4    = (base == 0) ? 1.0f : 0.0f;

#pragma unroll 1
  for (int t = 0; t < 12; ++t) {
    if (t < 11) __builtin_prefetch(stat + (rowBase + (long long)(t + 1) * 16) * 12, 0, 1);

    // A1 from global stat: lane row = t*16+Mrow, K = base..base+7 (K>=12 -> 0)
    const int rowA = t * 16 + Mrow;
    const float* rp = stat + (rowBase + rowA) * 12;
    const float4 g0 = *(const float4*)(rp + base);   // K base..base+3 (always valid)
    float4 g1 = *(const float4*)(rp + 4);            // K 4..7 (valid only if base==0)
    g1.x *= hi4; g1.y *= hi4; g1.z *= hi4; g1.w *= hi4;

    v16bf A1;
    A1[0] = (__bf16)g0.x; A1[1] = (__bf16)g0.y; A1[2] = (__bf16)g0.z; A1[3] = (__bf16)g0.w;
    A1[4] = (__bf16)g1.x; A1[5] = (__bf16)g1.y; A1[6] = (__bf16)g1.z; A1[7] = (__bf16)g1.w;
#pragma unroll
    for (int e = 8; e < 16; ++e) A1[e] = (__bf16)0.0f;

    v8f c0 = WMMA_BF16(A1, BW1, zero);               // es tile (cols 0..11)

    // stage (leaky(es+bias) | emb[phase]) into xs1 cols 0..15, branchless
#pragma unroll
    for (int r = 0; r < 8; ++r) {
      int Mr = r + rOff;
      int rw = t * 16 + Mr;
      float es = leaky(c0[r] + sb1v[b1Idx]) * esMask;
      float ev = sEmb[pBuf[wave][rw] * 4 + embCol] * (1.0f - esMask);
      xs1[wave][Mr][N] = (__bf16)(es + ev);
    }
    wave_fence();

    v16bf A2;
#pragma unroll
    for (int e = 0; e < 16; ++e) {
      int K = ((e >> 3) << 4) + base + (e & 7);
      A2[e] = xs1[wave][Mrow][K];
    }
    v8f d0 = WMMA_BF16(A2, BWe1a, zero);             // h cols 0..15
    v8f d1 = WMMA_BF16(A2, BWe1b, zero);             // h cols 16..31
#pragma unroll
    for (int r = 0; r < 8; ++r) {
      int Mr = r + rOff;
      xs2[wave][Mr][N]      = (__bf16)leaky(d0[r] + sbe1[N]);
      xs2[wave][Mr][16 + N] = (__bf16)leaky(d1[r] + sbe1[16 + N]);
    }
    wave_fence();

    v16bf A3;
#pragma unroll
    for (int e = 0; e < 16; ++e) {
      int K = ((e >> 3) << 4) + base + (e & 7);
      A3[e] = xs2[wave][Mrow][K];
    }
    v8f e0 = WMMA_BF16(A3, BWe2, zero);              // e tile (16 dims), K=32 exact
#pragma unroll
    for (int r = 0; r < 8; ++r) {
      int rw = t * 16 + r + rOff;
      eBuf[wave][rw][N] = (__bf16)(e0[r] + sbe2[N]);
    }
    wave_fence();                                    // covers xs1 reuse next tile
  }

  // ---- masked pooling: enc[b][p][d] = sum_m mf[p][m] * e[b][m][d] ----
#pragma unroll 1
  for (int j = 0; j < 64; ++j) {
    const int idx = j * 32 + lane;                   // 16b * 8p * 16d
    const int bl = idx >> 7;
    const int p  = (idx >> 4) & 7;
    const int d  = idx & 15;
    float acc = 0.f;
#pragma unroll
    for (int m = 0; m < BM; ++m)
      acc += sMask[p * BM + m] * (float)eBuf[wave][bl * BM + m][d];
    encB[wave][bl][p][d] = (__bf16)acc;
  }
  wave_fence();

  // =========== Stage B: pair-MLP via WMMA ===========
  const v16bf B1a = load_bfrag(Wc1, 20, 32, 0,  lane);
  const v16bf B1b = load_bfrag(Wc1, 20, 32, 16, lane);
  const v16bf B2a = load_bfrag(Wc2, 20, 20, 0,  lane);
  const v16bf B2b = load_bfrag(Wc2, 20, 20, 16, lane);
  const v16bf B3  = load_bfrag(Wc3, 1,  20, 0,  lane);
  const float bc3v = bc3[0];

  const float hiMask = (N < 4) ? 1.0f : 0.0f;        // layer outputs 16..19
  const int   hi1    = (N < 4) ? 16 + N : 19;        // clamped bias index

#pragma unroll 1
  for (int p = 0; p < PD; ++p) {
    v8f acc3 = zero;                                 // layer-3 out, summed over q
#pragma unroll 1
    for (int qi = 0; qi < PD - 1; ++qi) {
      const int q = (qi < p) ? qi : qi + 1;
      wave_fence();                                  // prior A3 reads vs new writes

      v16bf A1;                                      // x0 = concat(enc[p], enc[q])
#pragma unroll
      for (int e = 0; e < 16; ++e) {
        int K = ((e >> 3) << 4) + base + (e & 7);
        A1[e] = (K < 16) ? encB[wave][Mrow][p][K]
                         : encB[wave][Mrow][q][K - 16];
      }
      v8f c0 = WMMA_BF16(A1, B1a, zero);
      v8f c1 = WMMA_BF16(A1, B1b, zero);
#pragma unroll
      for (int r = 0; r < 8; ++r) {
        int Mr = r + rOff;
        xs1[wave][Mr][N]      = (__bf16)leaky(c0[r] + sbc1[N]);
        xs1[wave][Mr][16 + N] = (__bf16)(leaky(c1[r] + sbc1[hi1]) * hiMask);
      }
      wave_fence();

      v16bf A2;
#pragma unroll
      for (int e = 0; e < 16; ++e) {
        int K = ((e >> 3) << 4) + base + (e & 7);
        A2[e] = xs1[wave][Mrow][K];
      }
      v8f d0 = WMMA_BF16(A2, B2a, zero);
      v8f d1 = WMMA_BF16(A2, B2b, zero);
#pragma unroll
      for (int r = 0; r < 8; ++r) {
        int Mr = r + rOff;
        xs2[wave][Mr][N]      = (__bf16)leaky(d0[r] + sbc2[N]);
        xs2[wave][Mr][16 + N] = (__bf16)(leaky(d1[r] + sbc2[hi1]) * hiMask);
      }
      wave_fence();

      v16bf A3;
#pragma unroll
      for (int e = 0; e < 16; ++e) {
        int K = ((e >> 3) << 4) + base + (e & 7);
        A3[e] = xs2[wave][Mrow][K];
      }
      acc3 = WMMA_BF16(A3, B3, acc3);                // 20->1, accumulated over q
    }
    // Column 0 of acc3 holds sum_q x3; lanes N==0 own rows rOff..rOff+7.
    if (N == 0) {
#pragma unroll
      for (int r = 0; r < 8; ++r) {
        long long gb = bBase + r + rOff;
        out[gb * PD + p] = acc3[r] + 7.0f * bc3v;
      }
    }
  }
}

extern "C" void kernel_launch(void* const* d_in, const int* in_sizes, int n_in,
                              void* d_out, int out_size, void* d_ws, size_t ws_size,
                              hipStream_t stream) {
  (void)n_in; (void)d_ws; (void)ws_size; (void)out_size;
  const float* stat = (const float*)d_in[0];
  const int*   phase = (const int*)d_in[1];
  const int*   mask  = (const int*)d_in[2];
  const float* W1  = (const float*)d_in[3];
  const float* b1  = (const float*)d_in[4];
  const float* emb = (const float*)d_in[5];
  const float* We1 = (const float*)d_in[6];
  const float* be1 = (const float*)d_in[7];
  const float* We2 = (const float*)d_in[8];
  const float* be2 = (const float*)d_in[9];
  const float* Wc1 = (const float*)d_in[10];
  const float* bc1 = (const float*)d_in[11];
  const float* Wc2 = (const float*)d_in[12];
  const float* bc2 = (const float*)d_in[13];
  const float* Wc3 = (const float*)d_in[14];
  const float* bc3 = (const float*)d_in[15];
  float* out = (float*)d_out;

  const int Btot = in_sizes[0] / (BM * 12);   // 131072
  const int blocks = Btot / B_PER_BLOCK;      // 2048

  frap_fused_kernel<<<blocks, THREADS, 0, stream>>>(
      stat, phase, mask, W1, b1, emb, We1, be1, We2, be2,
      Wc1, bc1, Wc2, bc2, Wc3, bc3, out);
}
